// GCRN_58317065945462
// MI455X (gfx1250) — matrix operand order repaired and checked by reading, not compiled, not run on hardware
//
#include <hip/hip_runtime.h>
#include <hip/hip_bf16.h>
#include <stdint.h>

// ---------------- problem constants ----------------
#define B_DIM   8
#define N_NODES 1024
#define F_DIM   128
#define G_DIM   64
#define E_CAT   2
#define N_HEAD  4
#define N_EDGES 16384
#define L0_DIM  256
#define ALPHA   0.5f
#define BN_EPS  1e-5f
#define SLOPE   0.01f

#define COMB (B_DIM * E_CAT * N_HEAD)   // 64
#define ROWS (B_DIM * N_NODES)          // 8192
#define CATF (N_HEAD * E_CAT * G_DIM)   // 512 (head-major concat width)

typedef unsigned short u16;
typedef __attribute__((ext_vector_type(16))) __bf16 v16bf;
typedef __attribute__((ext_vector_type(8)))  float  v8f;
typedef int v4i __attribute__((vector_size(16)));   // int4 vector for async-LDS builtin

// Probe for the gfx1250 async global->LDS builtins (plain-copy fallback keeps
// compilation safe if this toolchain does not declare them).
#if defined(__has_builtin)
#if __has_builtin(__builtin_amdgcn_global_load_async_to_lds_b128)
#define HAVE_ASYNC_LDS 1
#endif
#endif
#ifndef HAVE_ASYNC_LDS
#define HAVE_ASYNC_LDS 0
#endif

static __device__ __forceinline__ void wait_asynccnt0() {
#if defined(__has_builtin) && __has_builtin(__builtin_amdgcn_s_wait_asynccnt)
  __builtin_amdgcn_s_wait_asynccnt(0);
#else
  asm volatile("s_wait_asynccnt 0x0" ::: "memory");
#endif
}

struct Frag32 { uint4 lo, hi; };   // 32 bytes == one v16bf fragment

// fp32 -> bf16 round-to-nearest-even
static __device__ __forceinline__ u16 f2bf(float x) {
  unsigned int u = __float_as_uint(x);
  u += 0x7FFFu + ((u >> 16) & 1u);
  return (u16)(u >> 16);
}

// ---------------- WMMA fragment helpers (wave32 layouts, CDNA5 ISA) ----
// A 16x32 bf16: lane<16 -> M=lane, K={k0..k0+7, k0+16..k0+23}
//               lane>=16 -> M=lane-16, K={k0+8..k0+15, k0+24..k0+31}
static __device__ __forceinline__ v16bf load_a_frag(const u16* A, int lda, int m0, int k0, int lane) {
  int row = m0 + (lane & 15);
  int kb  = k0 + ((lane >> 4) << 3);
  Frag32 f;
  f.lo = *(const uint4*)(A + (size_t)row * lda + kb);
  f.hi = *(const uint4*)(A + (size_t)row * lda + kb + 16);
  return __builtin_bit_cast(v16bf, f);
}
// B 32x16 bf16 with B stored as [Nc,K] (i.e. W[out,in], computing X @ W^T):
// lane<16 -> N=n0+lane, K={k0..k0+7, k0+16..k0+23} ; lane>=16 -> N=n0+lane-16, K={k0+8..k0+15, k0+24..k0+31}
static __device__ __forceinline__ v16bf load_b_frag(const u16* Bm, int ldb, int n0, int k0, int lane) {
  int col = n0 + (lane & 15);
  int kb  = k0 + ((lane >> 4) << 3);
  Frag32 f;
  f.lo = *(const uint4*)(Bm + (size_t)col * ldb + kb);
  f.hi = *(const uint4*)(Bm + (size_t)col * ldb + kb + 16);
  return __builtin_bit_cast(v16bf, f);
}
// C/D f32 16x16: VGPR r -> lanes 0-15 M=r, lanes 16-31 M=8+r; N = lane&15
static __device__ __forceinline__ void store_c(float* C, int ldc, int m0, int n0, int lane, v8f acc) {
  int col = n0 + (lane & 15);
  int rb  = m0 + ((lane >> 4) << 3);
#pragma unroll
  for (int r = 0; r < 8; ++r) C[(size_t)(rb + r) * ldc + col] = acc[r];
}

// one wave -> one 16x64 strip of C = A[M,K](bf16,row-major,lda=K) @ W[Nc,K]^T
// 4 accumulators: each A fragment feeds 4 WMMAs.
static __device__ __forceinline__ void gemm_strip(const u16* __restrict__ A,
                                                  const u16* __restrict__ Bm,
                                                  float* __restrict__ C,
                                                  int Nc, int K, int tm, int tn0, int lane) {
  v8f acc[4] = {{}, {}, {}, {}};
  for (int k0 = 0; k0 < K; k0 += 32) {
    if (k0 + 32 < K)
      __builtin_prefetch(A + (size_t)(tm + (lane & 15)) * K + k0 + 32, 0, 1);
    v16bf a = load_a_frag(A, K, tm, k0, lane);
#pragma unroll
    for (int j = 0; j < 4; ++j) {
      v16bf b = load_b_frag(Bm, K, tn0 + 16 * j, k0, lane);
      acc[j] = __builtin_amdgcn_wmma_f32_16x16x32_bf16(false, a, false, b,
                                                       (short)0, acc[j], false, false);
    }
  }
#pragma unroll
  for (int j = 0; j < 4; ++j) store_c(C, Nc, tm, tn0 + 16 * j, lane, acc[j]);
}

// ---------------- kernels ----------------

// flat fp32 -> bf16 convert (also used for W_emb1/W_lin0/W_lin1 which are already [out,in])
__global__ void k_conv_bf16(const float* __restrict__ in, u16* __restrict__ out, int n) {
  int i = blockIdx.x * blockDim.x + threadIdx.x;
  if (i < n) out[i] = f2bf(in[i]);
}

// Ws [E, H*F, G] -> WsT bf16 [E,H,G,F]  (so G is the "out" dim, contiguous over F)
__global__ void k_wst(const float* __restrict__ Ws, u16* __restrict__ WsT) {
  int i = blockIdx.x * blockDim.x + threadIdx.x;             // over E*H*G*F = 65536
  if (i >= E_CAT * N_HEAD * G_DIM * F_DIM) return;
  int f   = i % F_DIM;
  int g   = (i / F_DIM) % G_DIM;
  int eh  = i / (F_DIM * G_DIM);                             // e*H + h
  WsT[i] = f2bf(Ws[((size_t)eh * F_DIM + f) * G_DIM + g]);
}

// q[b,e,h,n] = X[b,n,:] . Wq_r[e,h,:]   (and same for v)
__global__ void k_qv(const float* __restrict__ X, const float* __restrict__ Wq,
                     const float* __restrict__ Wv, float* __restrict__ q, float* __restrict__ v) {
  int i = blockIdx.x * blockDim.x + threadIdx.x;             // over COMB*N = 65536
  if (i >= COMB * N_NODES) return;
  int n    = i % N_NODES;
  int comb = i / N_NODES;                                    // ((b*E+e)*H + h)
  int b    = comb / (E_CAT * N_HEAD);
  int eh   = comb % (E_CAT * N_HEAD);
  const float4* x4 = (const float4*)(X + ((size_t)b * N_NODES + n) * F_DIM);
  const float4* q4 = (const float4*)(Wq + (size_t)eh * F_DIM);
  const float4* v4 = (const float4*)(Wv + (size_t)eh * F_DIM);
  float sq = 0.f, sv = 0.f;
#pragma unroll 4
  for (int j = 0; j < F_DIM / 4; ++j) {
    float4 x = x4[j], a = q4[j], c = v4[j];
    sq += x.x * a.x + x.y * a.y + x.z * a.z + x.w * a.w;
    sv += x.x * c.x + x.y * c.y + x.z * c.z + x.w * c.w;
  }
  q[i] = sq; v[i] = sv;
}

// batched Wh = X[b] @ Ws_r[e,h] : [1024x128]@[128x64].
// The 16 KB B panel is staged to LDS once per block (async global->LDS on CDNA5),
// then 8 waves each compute a 16x64 strip reading B fragments from LDS.
__global__ void k_gemm_wh(const u16* __restrict__ Xbf, const u16* __restrict__ WsT,
                          float* __restrict__ Wh) {
  __shared__ __align__(16) u16 Bl[G_DIM * F_DIM];            // 64x128 bf16 = 16 KB
  int comb = blockIdx.y;                                     // ((b*E+e)*H + h)
  int b    = comb / (E_CAT * N_HEAD);
  int eh   = comb % (E_CAT * N_HEAD);
  const u16* Bsrc = WsT + (size_t)eh * G_DIM * F_DIM;

  // stage B panel: 4 passes x 256 threads x 16B
#pragma unroll
  for (int ph = 0; ph < (G_DIM * F_DIM) / (256 * 8); ++ph) {
    int e8 = (ph * 256 + threadIdx.x) * 8;
#if HAVE_ASYNC_LDS
    __builtin_amdgcn_global_load_async_to_lds_b128(
        (__attribute__((address_space(1))) v4i*)(Bsrc + e8),
        (__attribute__((address_space(3))) v4i*)(Bl + e8), 0, 0);
#else
    *(uint4*)(Bl + e8) = *(const uint4*)(Bsrc + e8);
#endif
  }
#if HAVE_ASYNC_LDS
  wait_asynccnt0();
#endif
  __syncthreads();

  int wave = threadIdx.x >> 5;
  int lane = threadIdx.x & 31;
  int tm   = (blockIdx.x * 8 + wave) << 4;                   // 64 strips cover N_NODES
  const u16* A = Xbf + (size_t)b * N_NODES * F_DIM;
  float*     C = Wh  + (size_t)comb * N_NODES * G_DIM;

  v8f acc[4] = {{}, {}, {}, {}};
  for (int k0 = 0; k0 < F_DIM; k0 += 32) {
    v16bf a = load_a_frag(A, F_DIM, tm, k0, lane);
#pragma unroll
    for (int j = 0; j < 4; ++j) {
      v16bf bf = load_b_frag(Bl, F_DIM, 16 * j, k0, lane);   // ds_load_b128 x2
      acc[j] = __builtin_amdgcn_wmma_f32_16x16x32_bf16(false, a, false, bf,
                                                       (short)0, acc[j], false, false);
    }
  }
#pragma unroll
  for (int j = 0; j < 4; ++j) store_c(C, G_DIM, tm, 16 * j, lane, acc[j]);
}

// generic flat GEMM: C[M,Nc] = A[M,K] @ W[Nc,K]^T, one 16x64 strip per wave
__global__ void k_gemm(const u16* __restrict__ A, const u16* __restrict__ Bm,
                       float* __restrict__ C, int M, int Nc, int K) {
  int wave = (blockIdx.x * blockDim.x + threadIdx.x) >> 5;
  int lane = threadIdx.x & 31;
  int tilesN = Nc >> 6;                                      // strips of 64 columns
  int tm  = (wave / tilesN) << 4;
  int tn0 = (wave % tilesN) << 6;
  if (tm >= M) return;
  gemm_strip(A, Bm, C, Nc, K, tm, tn0, lane);
}

// colWh[comb,g] = sum_n Wh[comb,n,g]
__global__ void k_colwh(const float* __restrict__ Wh, float* __restrict__ colWh) {
  int comb = blockIdx.x;
  int g    = threadIdx.x;                                    // 64 threads
  float s = 0.f;
  const float* base = Wh + (size_t)comb * N_NODES * G_DIM;
  for (int n = 0; n < N_NODES; ++n) s += base[(size_t)n * G_DIM + g];
  colWh[comb * G_DIM + g] = s;
}

// sparse softmax correction: for each edge (n<-m):
//   w = exp(leaky_relu(q_n * v_m)) - 1
//   Hacc[n,:] += w * Wh[m,:] ; denom[n] += w
__global__ void k_edges(const int* __restrict__ A, const float* __restrict__ q,
                        const float* __restrict__ v, const float* __restrict__ Wh,
                        float* __restrict__ Hacc, float* __restrict__ denom) {
  int idx = blockIdx.x;                                      // over B*E*N_EDGES
  int i   = idx % N_EDGES;
  int be  = idx / N_EDGES;                                   // b*E + e
  int g   = threadIdx.x;                                     // 0..63
  int base = (be * 2) * N_EDGES;
  int n = A[base + i];
  int m = A[base + N_EDGES + i];
#pragma unroll
  for (int h = 0; h < N_HEAD; ++h) {
    int comb = be * N_HEAD + h;
    float s = q[comb * N_NODES + n] * v[comb * N_NODES + m];
    s = (s >= 0.f) ? s : SLOPE * s;
    float w = __expf(s) - 1.f;
    atomicAdd(&Hacc[((size_t)comb * N_NODES + n) * G_DIM + g],
              w * Wh[((size_t)comb * N_NODES + m) * G_DIM + g]);
    if (g == 0) atomicAdd(&denom[comb * N_NODES + n], w);
  }
}

// H = (colWh + Hacc) / (N + denom), reordered to head-major concat [B*N, H*E*G] bf16
__global__ void k_combine(const float* __restrict__ Hacc, const float* __restrict__ colWh,
                          const float* __restrict__ denom, u16* __restrict__ Hcat) {
  size_t i = (size_t)blockIdx.x * blockDim.x + threadIdx.x;  // over COMB*N*G
  if (i >= (size_t)COMB * N_NODES * G_DIM) return;
  int g    = (int)(i % G_DIM);
  int n    = (int)((i / G_DIM) % N_NODES);
  int comb = (int)(i / ((size_t)G_DIM * N_NODES));
  int b = comb / (E_CAT * N_HEAD);
  int e = (comb / N_HEAD) % E_CAT;
  int h = comb % N_HEAD;
  float val = (colWh[comb * G_DIM + g] + Hacc[i]) /
              ((float)N_NODES + denom[comb * N_NODES + n]);
  Hcat[((size_t)b * N_NODES + n) * CATF + h * (E_CAT * G_DIM) + e * G_DIM + g] = f2bf(val);
}

// y = (1-alpha)*relu(H1) + alpha*X
__global__ void k_mix(const float* __restrict__ H1, const float* __restrict__ X,
                      float* __restrict__ y) {
  size_t i = (size_t)blockIdx.x * blockDim.x + threadIdx.x;  // over ROWS*F
  if (i >= (size_t)ROWS * F_DIM) return;
  float hv = H1[i];
  hv = hv > 0.f ? hv : 0.f;
  y[i] = (1.f - ALPHA) * hv + ALPHA * X[i];
}

// column-wise BN (training mode, biased var), optional residual add / ELU / bf16 out
__global__ void k_bn(const float* __restrict__ in, const float* __restrict__ resid,
                     float* __restrict__ outf, u16* __restrict__ outb,
                     int Rr, int Cc, int do_elu) {
  __shared__ float s1[256], s2[256];
  int c = blockIdx.x, t = threadIdx.x;
  float sum = 0.f, sq = 0.f;
  for (int r = t; r < Rr; r += 256) {
    float x = in[(size_t)r * Cc + c];
    if (resid) x += resid[(size_t)r * Cc + c];
    sum += x; sq += x * x;
  }
  s1[t] = sum; s2[t] = sq; __syncthreads();
  for (int s = 128; s > 0; s >>= 1) {
    if (t < s) { s1[t] += s1[t + s]; s2[t] += s2[t + s]; }
    __syncthreads();
  }
  float mu  = s1[0] / (float)Rr;
  float var = s2[0] / (float)Rr - mu * mu;
  float inv = rsqrtf(var + BN_EPS);
  for (int r = t; r < Rr; r += 256) {
    float x = in[(size_t)r * Cc + c];
    if (resid) x += resid[(size_t)r * Cc + c];
    float yv = (x - mu) * inv;
    if (do_elu) yv = (yv > 0.f) ? yv : (__expf(yv) - 1.f);
    if (outf) outf[(size_t)r * Cc + c] = yv;
    if (outb) outb[(size_t)r * Cc + c] = f2bf(yv);
  }
}

// ---------------- host ----------------
extern "C" void kernel_launch(void* const* d_in, const int* in_sizes, int n_in,
                              void* d_out, int out_size, void* d_ws, size_t ws_size,
                              hipStream_t stream) {
  (void)in_sizes; (void)n_in; (void)out_size; (void)ws_size;
  const int*   A     = (const int*)d_in[0];
  const float* X     = (const float*)d_in[1];
  const float* Ws    = (const float*)d_in[2];
  const float* Wq    = (const float*)d_in[3];
  const float* Wv    = (const float*)d_in[4];
  const float* Wemb1 = (const float*)d_in[5];
  const float* Wlin0 = (const float*)d_in[6];
  const float* Wlin1 = (const float*)d_in[7];
  float* Out = (float*)d_out;

  char* base = (char*)d_ws;
  size_t off = 0;
  auto alloc = [&](size_t bytes) -> void* {
    void* r = base + off;
    off += (bytes + 255) & ~(size_t)255;
    return r;
  };

  u16*   Xbf    = (u16*)  alloc((size_t)ROWS * F_DIM * 2);
  u16*   WsT    = (u16*)  alloc((size_t)E_CAT * N_HEAD * G_DIM * F_DIM * 2);
  u16*   We1b   = (u16*)  alloc((size_t)F_DIM * CATF * 2);
  u16*   Wl0b   = (u16*)  alloc((size_t)L0_DIM * F_DIM * 2);
  u16*   Wl1b   = (u16*)  alloc((size_t)F_DIM * L0_DIM * 2);
  float* qb     = (float*)alloc((size_t)COMB * N_NODES * 4);
  float* vb     = (float*)alloc((size_t)COMB * N_NODES * 4);
  float* Wh     = (float*)alloc((size_t)COMB * N_NODES * G_DIM * 4);
  float* colWh  = (float*)alloc((size_t)COMB * G_DIM * 4);
  float* Hacc   = (float*)alloc((size_t)COMB * N_NODES * G_DIM * 4);
  float* denom  = (float*)alloc((size_t)COMB * N_NODES * 4);
  u16*   Hcat   = (u16*)  alloc((size_t)ROWS * CATF * 2);
  float* H1     = (float*)alloc((size_t)ROWS * F_DIM * 4);
  float* yb     = (float*)alloc((size_t)ROWS * F_DIM * 4);
  float* ybn    = (float*)alloc((size_t)ROWS * F_DIM * 4);
  u16*   ybnb   = (u16*)  alloc((size_t)ROWS * F_DIM * 2);
  float* T0     = (float*)alloc((size_t)ROWS * L0_DIM * 4);
  u16*   Z1b    = (u16*)  alloc((size_t)ROWS * L0_DIM * 2);
  float* T1     = (float*)alloc((size_t)ROWS * F_DIM * 4);

  // --- prep: bf16 conversions ---
  k_conv_bf16<<<(ROWS * F_DIM + 255) / 256, 256, 0, stream>>>(X, Xbf, ROWS * F_DIM);
  k_conv_bf16<<<(F_DIM * CATF + 255) / 256, 256, 0, stream>>>(Wemb1, We1b, F_DIM * CATF);
  k_conv_bf16<<<(L0_DIM * F_DIM + 255) / 256, 256, 0, stream>>>(Wlin0, Wl0b, L0_DIM * F_DIM);
  k_conv_bf16<<<(F_DIM * L0_DIM + 255) / 256, 256, 0, stream>>>(Wlin1, Wl1b, F_DIM * L0_DIM);
  k_wst<<<(E_CAT * N_HEAD * G_DIM * F_DIM + 255) / 256, 256, 0, stream>>>(Ws, WsT);

  // --- q, v GEMVs ---
  k_qv<<<(COMB * N_NODES + 255) / 256, 256, 0, stream>>>(X, Wq, Wv, qb, vb);

  // --- Wh = X @ Ws_r : 64 batched [1024x128]@[128x64] WMMA GEMMs (LDS-staged B) ---
  {
    dim3 grid(8, COMB);            // 64 strips per combo, 8 waves per block
    k_gemm_wh<<<grid, 256, 0, stream>>>(Xbf, WsT, Wh);
  }
  k_colwh<<<COMB, G_DIM, 0, stream>>>(Wh, colWh);

  // --- sparse attention correction ---
  (void)hipMemsetAsync(Hacc, 0, (size_t)COMB * N_NODES * G_DIM * 4, stream);
  (void)hipMemsetAsync(denom, 0, (size_t)COMB * N_NODES * 4, stream);
  k_edges<<<B_DIM * E_CAT * N_EDGES, G_DIM, 0, stream>>>(A, qb, vb, Wh, Hacc, denom);
  {
    size_t tot = (size_t)COMB * N_NODES * G_DIM;
    k_combine<<<(unsigned)((tot + 255) / 256), 256, 0, stream>>>(Hacc, colWh, denom, Hcat);
  }

  // --- H1 = Hcat @ W_emb1^T : [8192x512]@[512x128] ---
  k_gemm<<<(ROWS / 16) * (F_DIM / 64) / 8, 256, 0, stream>>>(Hcat, We1b, H1, ROWS, F_DIM, CATF);
  k_mix<<<(ROWS * F_DIM + 255) / 256, 256, 0, stream>>>(H1, X, yb);
  k_bn<<<F_DIM, 256, 0, stream>>>(yb, nullptr, ybn, ybnb, ROWS, F_DIM, 0);

  // --- T0 = ybn @ W_lin0^T : [8192x128]@[128x256] ; BN + ELU ---
  k_gemm<<<(ROWS / 16) * (L0_DIM / 64) / 8, 256, 0, stream>>>(ybnb, Wl0b, T0, ROWS, L0_DIM, F_DIM);
  k_bn<<<L0_DIM, 256, 0, stream>>>(T0, nullptr, nullptr, Z1b, ROWS, L0_DIM, 1);

  // --- T1 = Z1 @ W_lin1^T : [8192x256]@[256x128] ; final BN(ybn + T1) -> out ---
  k_gemm<<<(ROWS / 16) * (F_DIM / 64) / 8, 256, 0, stream>>>(Z1b, Wl1b, T1, ROWS, F_DIM, L0_DIM);
  k_bn<<<F_DIM, 256, 0, stream>>>(T1, ybn, Out, nullptr, ROWS, F_DIM, 0);
}
